// GruAttentionDecoder_84507776516648
// MI455X (gfx1250) — matrix operand order repaired and checked by reading, not compiled, not run on hardware
//
#include <hip/hip_runtime.h>
#include <math.h>

// Problem constants (from the reference)
#define S_LEN 128
#define BATCH 32
#define E2DIM 1024
#define DHID  512
#define AHID  32
#define EMBD  256
#define VOCAB 32000
#define XDIM  (E2DIM + EMBD)   // 1280 = GRU input width
#define W1COL (DHID + E2DIM)   // 1536 = W1 row length
#define G3    (3 * DHID)       // 1536 = gate width

typedef __attribute__((ext_vector_type(16))) __bf16 v16bf;
typedef __attribute__((ext_vector_type(8)))  float  v8f;

__device__ __forceinline__ __bf16 f2bf(float f) {
  unsigned u = __builtin_bit_cast(unsigned, f);
  u += 0x7FFFu + ((u >> 16) & 1u);           // round-to-nearest-even
  unsigned short h = (unsigned short)(u >> 16);
  return __builtin_bit_cast(__bf16, h);
}

union Frag { v16bf v; float4 q[2]; };

// ---------------------------------------------------------------------------
// Shared M=32 bf16 WMMA GEMM body: C[32,N] = A[32,K] @ W[N,K]^T (+bias).
// One wave owns one 16-wide N tile (ntile) and both 16-row M tiles.
// K multiple of 32; N multiple of 16.
// A-frag per ISA 16-bit A 16x32 layout: lane<16 -> K{0..7,16..23},
// lane>=16 -> K{8..15,24..31}; both are two contiguous 16-byte loads from
// row-major A. B-frag: lane holds one N column, same K split as A.
// ---------------------------------------------------------------------------
__device__ __forceinline__ void gemm_m32_body(
    const __bf16* __restrict__ A,    // [32, K] row-major
    const __bf16* __restrict__ W,    // [N, K] row-major
    const float*  __restrict__ bias, // [N] or nullptr
    float* __restrict__ C,           // [32, N] row-major
    int N, int K, int ntile) {
  if (ntile * 16 >= N) return;
  const int lane = threadIdx.x & 31;
  const int ln   = lane & 15;
  const int half = lane >> 4;
  const int n    = ntile * 16 + ln;

  const __bf16* a0p = A + (size_t)ln * K        + half * 8;
  const __bf16* a1p = A + (size_t)(16 + ln) * K + half * 8;
  const __bf16* bp  = W + (size_t)n * K         + half * 8;

  v8f acc0 = {}; v8f acc1 = {};
  for (int k0 = 0; k0 < K; k0 += 32) {
    Frag a0, a1, b;
    a0.q[0] = *(const float4*)(a0p + k0);
    a0.q[1] = *(const float4*)(a0p + k0 + 16);
    a1.q[0] = *(const float4*)(a1p + k0);
    a1.q[1] = *(const float4*)(a1p + k0 + 16);
    b.q[0]  = *(const float4*)(bp  + k0);
    b.q[1]  = *(const float4*)(bp  + k0 + 16);
    acc0 = __builtin_amdgcn_wmma_f32_16x16x32_bf16(false, a0.v, false, b.v,
                                                   (short)0, acc0, false, false);
    acc1 = __builtin_amdgcn_wmma_f32_16x16x32_bf16(false, a1.v, false, b.v,
                                                   (short)0, acc1, false, false);
  }
  const float bv = bias ? bias[n] : 0.0f;
#pragma unroll
  for (int r = 0; r < 8; ++r) {
    const int m0 = half * 8 + r;                 // f32 C/D VGPR layout
    C[(size_t)m0 * N + n]        = acc0[r] + bv;
    C[(size_t)(16 + m0) * N + n] = acc1[r] + bv;
  }
}

// Standalone GEMM (used for the big logits matmul): 8 waves/block.
__global__ __launch_bounds__(256) void gemm_m32_bf16(
    const __bf16* __restrict__ A, const __bf16* __restrict__ W,
    const float* __restrict__ bias, float* __restrict__ C, int N, int K) {
  const int wave = threadIdx.x >> 5;
  gemm_m32_body(A, W, bias, C, N, K, blockIdx.x * 8 + wave);
}

// Fused GRU GEMM pair: blocks 0..11 compute gi = xbuf @ W_ih^T,
// blocks 12..23 compute gh = hbf @ W_hh^T. The two GEMMs are independent,
// so fusing them into one launch shortens the 48-step dependent chain and
// lets them run concurrently across WGPs.
__global__ __launch_bounds__(256) void gru_gemms(
    const __bf16* __restrict__ xbuf, const __bf16* __restrict__ hbf,
    const __bf16* __restrict__ Wih,  const __bf16* __restrict__ Whh,
    float* __restrict__ gi, float* __restrict__ gh) {
  const int wave = threadIdx.x >> 5;
  if (blockIdx.x < 12) {
    gemm_m32_body(xbuf, Wih, nullptr, gi, G3, XDIM, blockIdx.x * 8 + wave);
  } else {
    gemm_m32_body(hbf, Whh, nullptr, gh, G3, DHID, (blockIdx.x - 12) * 8 + wave);
  }
}

// ---------------------------------------------------------------------------
// One-time: fp32 -> bf16 weight conversion
// ---------------------------------------------------------------------------
__global__ void cvt_bf16(const float* __restrict__ src, __bf16* __restrict__ dst, int n) {
  int i = blockIdx.x * 256 + threadIdx.x;
  const int stride = gridDim.x * 256;
  for (; i < n; i += stride) dst[i] = f2bf(src[i]);
}

// ---------------------------------------------------------------------------
// One-time: encW1[(s*B+b)*AH + j] = b1[j] + enc[s,b,:] . W1[j, DH:]
// (the h-independent half of the attention MLP, hoisted out of the time loop)
// ---------------------------------------------------------------------------
__global__ __launch_bounds__(32) void encW1_kernel(
    const float* __restrict__ enc, const float* __restrict__ W1,
    const float* __restrict__ b1, float* __restrict__ encW1) {
  const int sb = blockIdx.x;           // s*BATCH + b
  const int j  = threadIdx.x;          // 0..31
  const float* e = enc + (size_t)sb * E2DIM;
  const float* w = W1 + (size_t)j * W1COL + DHID;
  float acc = b1[j];
  for (int k = 0; k < E2DIM; ++k) acc = fmaf(e[k], w[k], acc);
  encW1[(size_t)sb * AHID + j] = acc;
}

// ---------------------------------------------------------------------------
// One-time: h=0, h_bf16=0, xbuf embedding slot = emb[sos]
// ---------------------------------------------------------------------------
__global__ void init_state(const float* __restrict__ emb, const int* __restrict__ sos,
                           float* __restrict__ h, __bf16* __restrict__ hbf,
                           __bf16* __restrict__ xbuf) {
  const int i = blockIdx.x * 256 + threadIdx.x;
  if (i < BATCH * DHID) { h[i] = 0.0f; hbf[i] = __builtin_bit_cast(__bf16, (unsigned short)0); }
  if (i < BATCH * EMBD) {
    const int b = i >> 8, e = i & 255;
    xbuf[(size_t)b * XDIM + E2DIM + e] = f2bf(emb[(size_t)(*sos) * EMBD + e]);
  }
}

// ---------------------------------------------------------------------------
// Per step: attention.  One block per batch element, 128 threads (=S).
// hpart = h[b] @ W1h^T; score_s = W2 . relu(encW1[s,b]+hpart) + b2;
// softmax over S; context[b] = sum_s attn_s * enc[s,b]; write bf16 into xbuf.
// ---------------------------------------------------------------------------
__global__ __launch_bounds__(128) void attn_kernel(
    const float* __restrict__ enc, const float* __restrict__ encW1,
    const float* __restrict__ W1, const float* __restrict__ W2,
    const float* __restrict__ b2, const float* __restrict__ h,
    __bf16* __restrict__ xbuf) {
  const int b = blockIdx.x;
  const int t = threadIdx.x;
  __shared__ float hs[DHID];
  __shared__ float hpart[AHID];
  __shared__ float attn[S_LEN];
  __shared__ float red[S_LEN];

  for (int i = t; i < DHID; i += 128) hs[i] = h[(size_t)b * DHID + i];
  __syncthreads();

  if (t < AHID) {
    const float* w = W1 + (size_t)t * W1COL;   // h-columns of W1
    float acc = 0.0f;
    for (int k = 0; k < DHID; ++k) acc = fmaf(w[k], hs[k], acc);
    hpart[t] = acc;
  }
  __syncthreads();

  // scores (t == s)
  float sc = b2[0];
  {
    const float* ew = encW1 + ((size_t)t * BATCH + b) * AHID;
    for (int j = 0; j < AHID; ++j) {
      float v = ew[j] + hpart[j];
      v = v > 0.0f ? v : 0.0f;
      sc = fmaf(W2[j], v, sc);
    }
  }
  red[t] = sc;
  __syncthreads();
  for (int off = 64; off >= 1; off >>= 1) {
    if (t < off) red[t] = fmaxf(red[t], red[t + off]);
    __syncthreads();
  }
  const float mx = red[0];
  __syncthreads();
  const float ex = expf(sc - mx);
  attn[t] = ex; red[t] = ex;
  __syncthreads();
  for (int off = 64; off >= 1; off >>= 1) {
    if (t < off) red[t] += red[t + off];
    __syncthreads();
  }
  const float inv = 1.0f / red[0];
  __syncthreads();
  attn[t] *= inv;
  __syncthreads();

  // context: each thread owns 8 of the 1024 feature columns
  for (int e = t; e < E2DIM; e += 128) {
    float acc = 0.0f;
    for (int s = 0; s < S_LEN; ++s)
      acc = fmaf(enc[((size_t)s * BATCH + b) * E2DIM + e], attn[s], acc);
    xbuf[(size_t)b * XDIM + e] = f2bf(acc);
  }
}

// ---------------------------------------------------------------------------
// Per step: GRU gate math (torch order r,z,n); updates h (f32) and h_bf16.
// ---------------------------------------------------------------------------
__device__ __forceinline__ float sigmoidf_(float x) { return 1.0f / (1.0f + expf(-x)); }

__global__ void gates_kernel(const float* __restrict__ gi, const float* __restrict__ gh,
                             const float* __restrict__ b_ih, const float* __restrict__ b_hh,
                             float* __restrict__ h, __bf16* __restrict__ hbf) {
  const int i = blockIdx.x * 256 + threadIdx.x;
  if (i >= BATCH * DHID) return;
  const int b = i >> 9, d = i & (DHID - 1);
  const float* gib = gi + (size_t)b * G3;
  const float* ghb = gh + (size_t)b * G3;
  const float r = sigmoidf_(gib[d]           + b_ih[d]           + ghb[d]           + b_hh[d]);
  const float z = sigmoidf_(gib[DHID + d]    + b_ih[DHID + d]    + ghb[DHID + d]    + b_hh[DHID + d]);
  const float n = tanhf   (gib[2 * DHID + d] + b_ih[2 * DHID + d] +
                           r * (ghb[2 * DHID + d] + b_hh[2 * DHID + d]));
  const float hn = (1.0f - z) * n + z * h[i];
  h[i]   = hn;
  hbf[i] = f2bf(hn);
}

// ---------------------------------------------------------------------------
// Per step: argmax over V (first-max tie-break like jnp.argmax) + emb gather
// into the xbuf embedding slot (bf16) for the next step.
// ---------------------------------------------------------------------------
__global__ __launch_bounds__(256) void argmax_gather(
    const float* __restrict__ logits,   // [BATCH, VOCAB] for this step
    const float* __restrict__ emb, __bf16* __restrict__ xbuf) {
  const int b = blockIdx.x;
  const int t = threadIdx.x;
  __shared__ float bm[256];
  __shared__ int   bi[256];
  float best = -INFINITY; int besti = 0;
  for (int v = t; v < VOCAB; v += 256) {
    const float x = logits[(size_t)b * VOCAB + v];
    if (x > best) { best = x; besti = v; }
  }
  bm[t] = best; bi[t] = besti;
  __syncthreads();
  for (int off = 128; off >= 1; off >>= 1) {
    if (t < off) {
      if (bm[t + off] > bm[t] || (bm[t + off] == bm[t] && bi[t + off] < bi[t])) {
        bm[t] = bm[t + off]; bi[t] = bi[t + off];
      }
    }
    __syncthreads();
  }
  const int idx = bi[0];
  // EMB == 256 == blockDim
  xbuf[(size_t)b * XDIM + E2DIM + t] = f2bf(emb[(size_t)idx * EMBD + t]);
}

// ---------------------------------------------------------------------------
extern "C" void kernel_launch(void* const* d_in, const int* in_sizes, int n_in,
                              void* d_out, int out_size, void* d_ws, size_t ws_size,
                              hipStream_t stream) {
  const float* enc   = (const float*)d_in[0];
  // d_in[1] = max_length (device scalar; derived from out_size instead)
  const int*   sos   = (const int*)  d_in[2];
  const float* emb   = (const float*)d_in[3];
  const float* W1    = (const float*)d_in[4];
  const float* b1    = (const float*)d_in[5];
  const float* W2    = (const float*)d_in[6];
  const float* b2    = (const float*)d_in[7];
  const float* W_ih  = (const float*)d_in[8];
  const float* b_ih  = (const float*)d_in[9];
  const float* W_hh  = (const float*)d_in[10];
  const float* b_hh  = (const float*)d_in[11];
  const float* Wo    = (const float*)d_in[12];
  const float* bo    = (const float*)d_in[13];
  float* out = (float*)d_out;
  const int T = out_size / (BATCH * VOCAB);

  // workspace carve-up (256B aligned)
  char* p = (char*)d_ws;
  auto alloc = [&](size_t bytes) -> void* {
    void* r = (void*)p;
    p += (bytes + 255) & ~(size_t)255;
    return r;
  };
  __bf16* Wo_bf  = (__bf16*)alloc((size_t)VOCAB * DHID * 2);
  __bf16* Wih_bf = (__bf16*)alloc((size_t)G3 * XDIM * 2);
  __bf16* Whh_bf = (__bf16*)alloc((size_t)G3 * DHID * 2);
  float*  encW1  = (float*) alloc((size_t)S_LEN * BATCH * AHID * 4);
  float*  h      = (float*) alloc((size_t)BATCH * DHID * 4);
  __bf16* hbf    = (__bf16*)alloc((size_t)BATCH * DHID * 2);
  __bf16* xbuf   = (__bf16*)alloc((size_t)BATCH * XDIM * 2);
  float*  gi     = (float*) alloc((size_t)BATCH * G3 * 4);
  float*  gh     = (float*) alloc((size_t)BATCH * G3 * 4);

  // ---- one-time prep (all weights become L2-resident bf16) ----
  cvt_bf16<<<512, 256, 0, stream>>>(Wo,   Wo_bf,  VOCAB * DHID);
  cvt_bf16<<<512, 256, 0, stream>>>(W_ih, Wih_bf, G3 * XDIM);
  cvt_bf16<<<512, 256, 0, stream>>>(W_hh, Whh_bf, G3 * DHID);
  encW1_kernel<<<S_LEN * BATCH, 32, 0, stream>>>(enc, W1, b1, encW1);
  init_state<<<64, 256, 0, stream>>>(emb, sos, h, hbf, xbuf);

  // ---- sequential decode: 5 launches per step on the dependent chain ----
  const int blkLogits = (VOCAB / 16) / 8;   // 250
  for (int t = 0; t < T; ++t) {
    float* logits_t = out + (size_t)t * BATCH * VOCAB;
    attn_kernel<<<BATCH, 128, 0, stream>>>(enc, encW1, W1, W2, b2, h, xbuf);
    gru_gemms<<<24, 256, 0, stream>>>(xbuf, hbf, Wih_bf, Whh_bf, gi, gh);
    gates_kernel<<<64, 256, 0, stream>>>(gi, gh, b_ih, b_hh, h, hbf);
    gemm_m32_bf16<<<blkLogits, 256, 0, stream>>>(hbf, Wo_bf, bo, logits_t, VOCAB, DHID);
    argmax_gather<<<BATCH, 256, 0, stream>>>(logits_t, emb, xbuf);
  }
}
